// AttnBlock_53463752901034
// MI455X (gfx1250) — matrix operand order repaired and checked by reading, not compiled, not run on hardware
//
#include <hip/hip_runtime.h>

// ---------------------------------------------------------------------------
// Types for CDNA5 WMMA (wave32, v_wmma_f32_16x16x32_bf16) and TDM descriptors
// ---------------------------------------------------------------------------
typedef __attribute__((ext_vector_type(8)))  float    v8f;
typedef __attribute__((ext_vector_type(8)))  __bf16   v8bf;
typedef __attribute__((ext_vector_type(16))) __bf16   v16bf;
typedef __attribute__((ext_vector_type(4)))  unsigned v4u;
typedef __attribute__((ext_vector_type(4)))  int      v4i;
typedef __attribute__((ext_vector_type(8)))  int      v8i;

#define ATTN_SCALE 0.125f   // (DIM/HEADS)^-0.5 = 64^-0.5
#define LN_EPS 1e-5f

__device__ __forceinline__ v16bf cat16(v8bf lo, v8bf hi) {
  return __builtin_shufflevector(lo, hi, 0,1,2,3,4,5,6,7,8,9,10,11,12,13,14,15);
}

// A fragment: 16x32 bf16, source row-major [m][k] with leading dim ld.
__device__ __forceinline__ v16bf frag_a(const __bf16* tile, int ld) {
  int lane = threadIdx.x & 31;
  int m = lane & 15, hf = lane >> 4;
  const __bf16* p = tile + m * ld + 8 * hf;
  v8bf lo = *(const v8bf*)(p);
  v8bf hi = *(const v8bf*)(p + 16);
  return cat16(lo, hi);
}

// B fragment: 32x16 bf16, source stored TRANSPOSED row-major [n][k].
__device__ __forceinline__ v16bf frag_b(const __bf16* tileT, int ld) {
  int lane = threadIdx.x & 31;
  int n = lane & 15, hf = lane >> 4;
  const __bf16* p = tileT + n * ld + 8 * hf;
  v8bf lo = *(const v8bf*)(p);
  v8bf hi = *(const v8bf*)(p + 16);
  return cat16(lo, hi);
}

__device__ __forceinline__ v8f wmma_bf16(v16bf a, v16bf b, v8f c) {
  return __builtin_amdgcn_wmma_f32_16x16x32_bf16(false, a, false, b, (short)0, c,
                                                 false, false);
}

__device__ __forceinline__ float gelu_f(float x) {
  const float c0 = 0.7978845608028654f;  // sqrt(2/pi)
  return 0.5f * x * (1.0f + tanhf(c0 * (x + 0.044715f * x * x * x)));
}

// ---------------------------------------------------------------------------
// Tensor Data Mover: 2D strided tile (rows x 32 bf16) -> contiguous LDS.
// D# per CDNA5 ISA ch.8: group0 {count, lds_addr, global_addr, type=2},
// group1 {data_size=1(2B), tensor_dim0/1, tile_dim0/1, tensor_dim0_stride}.
// Issued by one wave; completion via s_wait_tensorcnt.
// ---------------------------------------------------------------------------
__device__ __forceinline__ void tdm_load_2d(unsigned lds_off, const void* gptr,
                                            unsigned tile_k, unsigned tile_n,
                                            unsigned tensor_k, unsigned tensor_n,
                                            unsigned stride_elems) {
  unsigned long long ga = (unsigned long long)(size_t)gptr;
  v4u g0;
  g0[0] = 1u;                                   // count=1, no gather/restore
  g0[1] = lds_off;                              // lds_addr (bytes)
  g0[2] = (unsigned)(ga & 0xFFFFFFFFu);         // global_addr[31:0]
  g0[3] = (unsigned)((ga >> 32) & 0x1FFFFFFu)   // global_addr[56:32]
          | (2u << 30);                         // type=2 ("image")
  v8i g1;
  g1[0] = (int)(1u << 16);                                   // data_size=1 (2B)
  g1[1] = (int)((tensor_k & 0xFFFFu) << 16);                 // tensor_dim0 lo
  g1[2] = (int)(((tensor_k >> 16) & 0xFFFFu) |
                ((tensor_n & 0xFFFFu) << 16));               // dim0 hi | dim1 lo
  g1[3] = (int)(((tensor_n >> 16) & 0xFFFFu) |
                ((tile_k & 0xFFFFu) << 16));                 // dim1 hi | tile_dim0
  g1[4] = (int)(tile_n & 0xFFFFu);                           // tile_dim1, tile_dim2=0
  g1[5] = (int)stride_elems;                                 // tensor_dim0_stride lo
  g1[6] = 0;
  g1[7] = 0;
  v4i z4 = {0, 0, 0, 0};
#if __clang_major__ >= 23
  v8i z8 = {0, 0, 0, 0, 0, 0, 0, 0};
  __builtin_amdgcn_tensor_load_to_lds(g0, g1, z4, z4, z8, 0);
#else
  __builtin_amdgcn_tensor_load_to_lds(g0, g1, z4, z4, 0);
#endif
}

// ---------------------------------------------------------------------------
// f32 -> bf16 cast
// ---------------------------------------------------------------------------
__global__ void cast_f32_bf16(const float* __restrict__ in,
                              __bf16* __restrict__ out, int n) {
  int i = blockIdx.x * blockDim.x + threadIdx.x;
  if (i < n) out[i] = (__bf16)in[i];
}

// transpose + cast: in[R][C] f32 -> out[C][R] bf16  (weights, done once)
__global__ void transpose_cast(const float* __restrict__ in,
                               __bf16* __restrict__ out, int R, int C) {
  int i = blockIdx.x * blockDim.x + threadIdx.x;
  if (i < R * C) {
    int r = i / C, c = i - r * C;
    out[(size_t)c * R + r] = (__bf16)in[i];
  }
}

// ---------------------------------------------------------------------------
// WMMA GEMM: C[M,N] = op(A[M,K] @ Bt[N,K]^T + bias), Bt pre-transposed.
// Block: 256 threads (8 waves). Tile: BM=128, BN=64, BK=32.
// Both LDS tiles staged by the Tensor Data Mover (wave 0 issues descriptors).
// ---------------------------------------------------------------------------
template <bool GELU, bool OUT_BF16>
__global__ void gemm_wmma(const __bf16* __restrict__ A,
                          const __bf16* __restrict__ Bt,
                          const float* __restrict__ bias, void* __restrict__ Cout,
                          int M, int N, int K) {
  __shared__ __bf16 sA[128 * 32];   // [m][k]
  __shared__ __bf16 sBt[64 * 32];   // [n][k]
  int tid = threadIdx.x;
  int m0 = blockIdx.y * 128;
  int n0 = blockIdx.x * 64;
  int wave = tid >> 5;

  v8f acc[4] = {};

  for (int k0 = 0; k0 < K; k0 += 32) {
    __syncthreads();  // previous iteration's readers done before TDM overwrites
    if (wave == 0) {
      tdm_load_2d((unsigned)(size_t)(const void*)sA,
                  A + (size_t)m0 * K + k0, 32, 128, (unsigned)K, (unsigned)M,
                  (unsigned)K);
      tdm_load_2d((unsigned)(size_t)(const void*)sBt,
                  Bt + (size_t)n0 * K + k0, 32, 64, (unsigned)K, (unsigned)N,
                  (unsigned)K);
      __builtin_amdgcn_s_wait_tensorcnt(0);
    } else if (k0 + 32 < K) {
      // hint next A tile toward L2 while TDM runs
      __builtin_prefetch(A + (size_t)(m0 + (tid >> 1)) * K + k0 + 32, 0, 1);
    }
    __syncthreads();

    v16bf a = frag_a(sA + wave * 16 * 32, 32);
#pragma unroll
    for (int nt = 0; nt < 4; ++nt) {
      v16bf bb = frag_b(sBt + nt * 16 * 32, 32);
      acc[nt] = wmma_bf16(a, bb, acc[nt]);
    }
  }

  // Epilogue. C/D layout: VGPR r -> row (r + 8*half), col = lane&15.
  int lane = tid & 31, hf = lane >> 4, n = lane & 15;
#pragma unroll
  for (int nt = 0; nt < 4; ++nt) {
    int col = n0 + nt * 16 + n;
    float bv = bias ? bias[col] : 0.0f;
#pragma unroll
    for (int r = 0; r < 8; ++r) {
      int row = m0 + wave * 16 + r + 8 * hf;
      float v = acc[nt][r] + bv;
      if (GELU) v = gelu_f(v);
      if (OUT_BF16)
        ((__bf16*)Cout)[(size_t)row * N + col] = (__bf16)v;
      else
        ((float*)Cout)[(size_t)row * N + col] = v;
    }
  }
}

// ---------------------------------------------------------------------------
// Column softmax stats for k (softmax over sequence dim).
// ---------------------------------------------------------------------------
__global__ void softmax_stats(const __bf16* __restrict__ qkv,
                              float* __restrict__ colmax,
                              float* __restrict__ rcpsum) {
  int c = blockIdx.x;
  int bb = c >> 8, rem = c & 255;
  const __bf16* base = qkv + (size_t)bb * 4096 * 768 + 256 + rem;
  __shared__ float red[256];
  int t = threadIdx.x;

  float m = -3.0e38f;
  for (int i = t; i < 4096; i += 256) m = fmaxf(m, (float)base[(size_t)i * 768]);
  red[t] = m;
  __syncthreads();
  for (int s = 128; s > 0; s >>= 1) {
    if (t < s) red[t] = fmaxf(red[t], red[t + s]);
    __syncthreads();
  }
  m = red[0];
  __syncthreads();

  float sum = 0.0f;
  for (int i = t; i < 4096; i += 256)
    sum += __expf((float)base[(size_t)i * 768] - m);
  red[t] = sum;
  __syncthreads();
  for (int s = 128; s > 0; s >>= 1) {
    if (t < s) red[t] += red[t + s];
    __syncthreads();
  }
  if (t == 0) {
    colmax[c] = m;
    rcpsum[c] = 1.0f / red[0];
  }
}

// ---------------------------------------------------------------------------
// build_m: per (b,h): M[d,d'] = sum_i exp(k[i,d]-max[d]) * (q[i,d']*scale),
// row d scaled by rcpsum[d]; stored TRANSPOSED bf16 for direct B-frag use.
// ---------------------------------------------------------------------------
__global__ void build_m(const __bf16* __restrict__ qkv,
                        const float* __restrict__ colmax,
                        const float* __restrict__ rcpsum,
                        __bf16* __restrict__ MT) {
  int g = blockIdx.x;  // 0..7
  int bb = g >> 2, hh = g & 3;
  const __bf16* kbase = qkv + (size_t)bb * 4096 * 768 + 256 + hh * 64;
  const __bf16* qbase = qkv + (size_t)bb * 4096 * 768 + hh * 64;

  __shared__ __bf16 ldsK[64 * 32];  // [d][i]
  __shared__ __bf16 ldsQ[64 * 32];  // [d'][i]

  int tid = threadIdx.x;
  int wave = tid >> 5;
  int mt = wave & 3;
  int np2 = (wave >> 2) * 2;
  int si = tid >> 3;
  int ds = (tid & 7) * 8;

  float cm[8];
#pragma unroll
  for (int j = 0; j < 8; ++j) cm[j] = colmax[g * 64 + ds + j];

  v8f acc0 = {}, acc1 = {};

  for (int i0 = 0; i0 < 4096; i0 += 32) {
    __syncthreads();
    v8bf kv = *(const v8bf*)(kbase + (size_t)(i0 + si) * 768 + ds);
    v8bf qv = *(const v8bf*)(qbase + (size_t)(i0 + si) * 768 + ds);
#pragma unroll
    for (int j = 0; j < 8; ++j) {
      ldsK[(ds + j) * 32 + si] = (__bf16)__expf((float)kv[j] - cm[j]);
      ldsQ[(ds + j) * 32 + si] = (__bf16)((float)qv[j] * ATTN_SCALE);
    }
    __syncthreads();
    v16bf a = frag_a(ldsK + mt * 16 * 32, 32);
    v16bf b0 = frag_b(ldsQ + (np2 + 0) * 16 * 32, 32);
    v16bf b1 = frag_b(ldsQ + (np2 + 1) * 16 * 32, 32);
    acc0 = wmma_bf16(a, b0, acc0);
    acc1 = wmma_bf16(a, b1, acc1);
  }

  int lane = tid & 31, hf = lane >> 4, n = lane & 15;
#pragma unroll
  for (int r = 0; r < 8; ++r) {
    int d = mt * 16 + r + 8 * hf;
    float rs = rcpsum[g * 64 + d];
    int d0 = (np2 + 0) * 16 + n;
    int d1 = (np2 + 1) * 16 + n;
    MT[(size_t)g * 4096 + d0 * 64 + d] = (__bf16)(acc0[r] * rs);
    MT[(size_t)g * 4096 + d1 * 64 + d] = (__bf16)(acc1[r] * rs);
  }
}

// ---------------------------------------------------------------------------
// apply_m: out[b,h,i,d'] = sum_d v[i,d] * Mn[d,d'], with the reference's
// "faithful reshape" scramble into attn_in[b, token, c] as bf16.
// ---------------------------------------------------------------------------
__global__ void apply_m(const __bf16* __restrict__ qkv,
                        const __bf16* __restrict__ MT,
                        __bf16* __restrict__ attn_in) {
  int g = blockIdx.y;
  int bb = g >> 2, hh = g & 3;
  int i0 = blockIdx.x * 128;
  const __bf16* vbase = qkv + (size_t)bb * 4096 * 768 + 512 + hh * 64;
  const __bf16* MTg = MT + (size_t)g * 4096;

  __shared__ __bf16 ldsV[128 * 64];   // [i][d]
  __shared__ __bf16 ldsMT[64 * 64];   // [d'][d]
  int tid = threadIdx.x;

#pragma unroll
  for (int rep = 0; rep < 4; ++rep) {
    int r = (tid >> 3) + rep * 32;
    int ds = (tid & 7) * 8;
    v8bf x = *(const v8bf*)(vbase + (size_t)(i0 + r) * 768 + ds);
    *(v8bf*)(&ldsV[r * 64 + ds]) = x;
  }
  {
    int r = tid >> 2;
    int s = (tid & 3) * 16;
    v8bf x0 = *(const v8bf*)(MTg + r * 64 + s);
    v8bf x1 = *(const v8bf*)(MTg + r * 64 + s + 8);
    *(v8bf*)(&ldsMT[r * 64 + s]) = x0;
    *(v8bf*)(&ldsMT[r * 64 + s + 8]) = x1;
  }
  __syncthreads();

  int wave = tid >> 5;
  const __bf16* arow = ldsV + wave * 16 * 64;
  v16bf a0 = frag_a(arow + 0, 64);
  v16bf a1 = frag_a(arow + 32, 64);
  v8f acc[4] = {};
#pragma unroll
  for (int nt = 0; nt < 4; ++nt) {
    v16bf b0 = frag_b(ldsMT + nt * 16 * 64 + 0, 64);
    v16bf b1 = frag_b(ldsMT + nt * 16 * 64 + 32, 64);
    acc[nt] = wmma_bf16(a0, b0, acc[nt]);
    acc[nt] = wmma_bf16(a1, b1, acc[nt]);
  }

  int lane = tid & 31, hf = lane >> 4, n = lane & 15;
#pragma unroll
  for (int nt = 0; nt < 4; ++nt) {
#pragma unroll
    for (int r = 0; r < 8; ++r) {
      int i = i0 + wave * 16 + r + 8 * hf;
      int dp = nt * 16 + n;
      long flat = ((long)hh * 4096 + i) * 64 + dp;  // within batch b
      int token = (int)(flat >> 8);
      int cc = (int)(flat & 255);
      attn_in[((size_t)bb * 4096 + token) * 256 + cc] = (__bf16)acc[nt][r];
    }
  }
}

// ---------------------------------------------------------------------------
// LayerNorm helpers (row length 256, one block per row, 256 threads)
// ---------------------------------------------------------------------------
__device__ __forceinline__ float block_ln(float v, const float* g, const float* b,
                                          float* red, int c) {
  red[c] = v;
  __syncthreads();
  for (int s = 128; s > 0; s >>= 1) {
    if (c < s) red[c] += red[c + s];
    __syncthreads();
  }
  float mu = red[0] * (1.0f / 256.0f);
  __syncthreads();
  float d = v - mu;
  red[c] = d * d;
  __syncthreads();
  for (int s = 128; s > 0; s >>= 1) {
    if (c < s) red[c] += red[c + s];
    __syncthreads();
  }
  float var = red[0] * (1.0f / 256.0f);
  __syncthreads();
  return d * rsqrtf(var + LN_EPS) * g[c] + b[c];
}

__global__ void ln_attn(const float* __restrict__ y, const float* __restrict__ x,
                        const float* __restrict__ g_ao, const float* __restrict__ be_ao,
                        const float* __restrict__ g1, const float* __restrict__ be1,
                        float* __restrict__ h1, __bf16* __restrict__ h1bf) {
  __shared__ float red[256];
  int row = blockIdx.x, c = threadIdx.x;
  size_t idx = (size_t)row * 256 + c;
  float t = y[idx];
  t = block_ln(t, g_ao, be_ao, red, c);
  t = block_ln(t, g1, be1, red, c);
  float o = x[idx] + t;
  h1[idx] = o;
  h1bf[idx] = (__bf16)o;
}

__global__ void ln_final(const float* __restrict__ mlp2, const float* __restrict__ x,
                         const float* __restrict__ h1,
                         const float* __restrict__ g2, const float* __restrict__ be2,
                         float* __restrict__ out) {
  __shared__ float red[256];
  int row = blockIdx.x, c = threadIdx.x;
  size_t idx = (size_t)row * 256 + c;
  float t = block_ln(mlp2[idx], g2, be2, red, c);
  out[idx] = x[idx] + h1[idx] + t;
}

// ---------------------------------------------------------------------------
// Host launch
// ---------------------------------------------------------------------------
extern "C" void kernel_launch(void* const* d_in, const int* in_sizes, int n_in,
                              void* d_out, int out_size, void* d_ws, size_t ws_size,
                              hipStream_t stream) {
  const float* x     = (const float*)d_in[0];
  const float* wqkv  = (const float*)d_in[1];
  const float* wao   = (const float*)d_in[2];
  const float* b_ao  = (const float*)d_in[3];
  const float* g_ao  = (const float*)d_in[4];
  const float* be_ao = (const float*)d_in[5];
  const float* g1    = (const float*)d_in[6];
  const float* be1   = (const float*)d_in[7];
  const float* wfc1  = (const float*)d_in[8];
  const float* b_fc1 = (const float*)d_in[9];
  const float* wfc2  = (const float*)d_in[10];
  const float* b_fc2 = (const float*)d_in[11];
  const float* g2    = (const float*)d_in[12];
  const float* be2   = (const float*)d_in[13];
  float* out = (float*)d_out;

  const int M = 8192;  // 2 * 4096 tokens

  char* p = (char*)d_ws;
  auto alloc = [&](size_t bytes) -> void* {
    void* r = (void*)p;
    p += (bytes + 255) & ~(size_t)255;
    return r;
  };
  __bf16* xbf     = (__bf16*)alloc((size_t)M * 256 * 2);
  __bf16* wqkvT   = (__bf16*)alloc((size_t)256 * 768 * 2);   // [768][256]
  __bf16* waoT    = (__bf16*)alloc((size_t)256 * 256 * 2);   // [256][256]
  __bf16* wfc1T   = (__bf16*)alloc((size_t)256 * 1024 * 2);  // [1024][256]
  __bf16* wfc2T   = (__bf16*)alloc((size_t)1024 * 256 * 2);  // [256][1024]
  __bf16* qkvbf   = (__bf16*)alloc((size_t)M * 768 * 2);
  float*  colmax  = (float*)alloc(512 * 4);
  float*  rcpsum  = (float*)alloc(512 * 4);
  __bf16* MT      = (__bf16*)alloc((size_t)8 * 64 * 64 * 2);
  __bf16* attn_in = (__bf16*)alloc((size_t)M * 256 * 2);
  float*  yattn   = (float*)alloc((size_t)M * 256 * 4);
  float*  h1      = (float*)alloc((size_t)M * 256 * 4);
  __bf16* h1bf    = (__bf16*)alloc((size_t)M * 256 * 2);
  __bf16* mlp1    = (__bf16*)alloc((size_t)M * 1024 * 2);
  float*  mlp2    = (float*)alloc((size_t)M * 256 * 4);

  // 1. casts / weight transposes (weights become Bt[N][K] bf16)
  cast_f32_bf16<<<(M * 256) / 256, 256, 0, stream>>>(x, xbf, M * 256);
  transpose_cast<<<(256 * 768) / 256, 256, 0, stream>>>(wqkv, wqkvT, 256, 768);
  transpose_cast<<<(256 * 256) / 256, 256, 0, stream>>>(wao, waoT, 256, 256);
  transpose_cast<<<(256 * 1024) / 256, 256, 0, stream>>>(wfc1, wfc1T, 256, 1024);
  transpose_cast<<<(1024 * 256) / 256, 256, 0, stream>>>(wfc2, wfc2T, 1024, 256);

  // 2. qkv = x @ w_qkv   [8192,256]x[256,768] -> bf16
  gemm_wmma<false, true><<<dim3(768 / 64, M / 128), 256, 0, stream>>>(
      xbf, wqkvT, (const float*)nullptr, (void*)qkvbf, M, 768, 256);

  // 3. softmax stats over sequence dim for k
  softmax_stats<<<512, 256, 0, stream>>>(qkvbf, colmax, rcpsum);

  // 4. M = softmax(k)^T (q*scale) per (b,h), stored transposed bf16
  build_m<<<8, 256, 0, stream>>>(qkvbf, colmax, rcpsum, MT);

  // 5. out = v @ M with reshape-scramble epilogue -> attn_in bf16 [8192,256]
  apply_m<<<dim3(4096 / 128, 8), 256, 0, stream>>>(qkvbf, MT, attn_in);

  // 6. yattn = attn_in @ w_ao + b_ao (f32), then double-LN residual -> h1
  gemm_wmma<false, false><<<dim3(256 / 64, M / 128), 256, 0, stream>>>(
      attn_in, waoT, b_ao, (void*)yattn, M, 256, 256);
  ln_attn<<<M, 256, 0, stream>>>(yattn, x, g_ao, be_ao, g1, be1, h1, h1bf);

  // 7. mlp1 = gelu(h1 @ w_fc1 + b_fc1) -> bf16 [8192,1024]
  gemm_wmma<true, true><<<dim3(1024 / 64, M / 128), 256, 0, stream>>>(
      h1bf, wfc1T, b_fc1, (void*)mlp1, M, 1024, 256);

  // 8. mlp2 = mlp1 @ w_fc2 + b_fc2 -> f32, then final LN + residuals
  gemm_wmma<false, false><<<dim3(256 / 64, M / 128), 256, 0, stream>>>(
      mlp1, wfc2T, b_fc2, (void*)mlp2, M, 256, 1024);
  ln_final<<<M, 256, 0, stream>>>(mlp2, x, h1, g2, be2, out);

  (void)in_sizes; (void)n_in; (void)out_size; (void)ws_size;
}
